// Model_82205674045593
// MI455X (gfx1250) — matrix operand order repaired and checked by reading, not compiled, not run on hardware
//
#include <hip/hip_runtime.h>

// ---------------------------------------------------------------------------
// Problem constants: B=256, T=128, D=768, K=74 (padded to KP=96), C=128
// ---------------------------------------------------------------------------
#define NB   256
#define NT   128
#define ND   768
#define NK   74
#define KP   96              // padded K (multiple of 32; pad cols are zero)
#define NC   128
#define NTKP (NT * KP)       // 12288 padded per-batch flattened length

typedef __attribute__((ext_vector_type(16))) __bf16 v16bf;
typedef __attribute__((ext_vector_type(8))) unsigned int v8u;
typedef __attribute__((ext_vector_type(8))) float v8f;

__device__ __forceinline__ unsigned int pack2bf(float x0, float x1) {
  // two fp32 -> packed bf16 pair (round-to-nearest-even), x0 in low half
  unsigned int u0 = __float_as_uint(x0);
  unsigned int u1 = __float_as_uint(x1);
  u0 = (u0 + 0x7FFFu + ((u0 >> 16) & 1u)) >> 16;
  u1 = (u1 + 0x7FFFu + ((u1 >> 16) & 1u)) & 0xFFFF0000u;
  return u0 | u1;
}

#define WMMA_BF16(a, b, c)                                                \
  __builtin_amdgcn_wmma_f32_16x16x32_bf16(false, __builtin_bit_cast(v16bf, (a)), \
                                          false, __builtin_bit_cast(v16bf, (b)), \
                                          (short)0, (c), false, false)

__device__ __forceinline__ v8u frag_from_uint4(uint4 lo, uint4 hi) {
  v8u u;
  u[0] = lo.x; u[1] = lo.y; u[2] = lo.z; u[3] = lo.w;
  u[4] = hi.x; u[5] = hi.y; u[6] = hi.z; u[7] = hi.w;
  return u;
}

// --- CDNA5 async global->LDS copy (ASYNCcnt-tracked; no VGPR round trip) ---
// LDS aperture maps addr[31:0] to the LDS offset, so the low 32 bits of a
// flat __shared__ pointer are the LDS byte address the instruction expects.
__device__ __forceinline__ void async_g2l_b128(void* lds, const void* g) {
  unsigned loff = (unsigned)(size_t)lds;
  asm volatile("global_load_async_to_lds_b128 %0, %1, off"
               :: "v"(loff), "v"(g) : "memory");
}
__device__ __forceinline__ void wait_async0() {
  asm volatile("s_wait_asynccnt 0" ::: "memory");
}

// A fragment (16x32 bf16) from row-major fp32, NO bounds (ld, k0 32-aligned).
__device__ __forceinline__ v8u load_a_frag(const float* __restrict__ src, int ld,
                                           int row0, int k0) {
  const int lane = threadIdx.x & 31;
  const int half = lane >> 4;
  const int m = lane & 15;
  const float* rp = src + (size_t)(row0 + m) * ld + k0 + half * 16;
  const float4 a0 = ((const float4*)rp)[0];
  const float4 a1 = ((const float4*)rp)[1];
  const float4 b0 = ((const float4*)(rp + 8))[0];
  const float4 b1 = ((const float4*)(rp + 8))[1];
  v8u u;
  u[0] = pack2bf(a0.x, a0.y); u[1] = pack2bf(a0.z, a0.w);
  u[2] = pack2bf(a1.x, a1.y); u[3] = pack2bf(a1.z, a1.w);
  u[4] = pack2bf(b0.x, b0.y); u[5] = pack2bf(b0.z, b0.w);
  u[6] = pack2bf(b1.x, b1.y); u[7] = pack2bf(b1.z, b1.w);
  return u;
}

// B fragment where logical B[k][n] = src[(n0+n)*ld + k] (contiguous in k)
__device__ __forceinline__ v8u load_bt_frag(const float* __restrict__ src, int ld,
                                            int k0, int n0) {
  const int lane = threadIdx.x & 31;
  const int half = lane >> 4;
  const int n = lane & 15;
  const float* cp = src + (size_t)(n0 + n) * ld + k0 + half * 16;
  const float4 c0 = ((const float4*)cp)[0];
  const float4 c1 = ((const float4*)cp)[1];
  const float4 c2 = ((const float4*)cp)[2];
  const float4 c3 = ((const float4*)cp)[3];
  v8u u;
  u[0] = pack2bf(c0.x, c0.y); u[1] = pack2bf(c0.z, c0.w);
  u[2] = pack2bf(c1.x, c1.y); u[3] = pack2bf(c1.z, c1.w);
  u[4] = pack2bf(c2.x, c2.y); u[5] = pack2bf(c2.z, c2.w);
  u[6] = pack2bf(c3.x, c3.y); u[7] = pack2bf(c3.z, c3.w);
  return u;
}

// B fragment (32x16) from row-major src[k*ld + n], k-strided, no bounds.
__device__ __forceinline__ v8u load_b_frag(const float* __restrict__ src, int ld,
                                           int k0, int n0) {
  const int lane = threadIdx.x & 31;
  const int half = lane >> 4;
  const int col = n0 + (lane & 15);
  const float* p = src + (size_t)(k0 + half * 16) * ld + col;
  v8u u;
#pragma unroll
  for (int v = 0; v < 8; ++v)
    u[v] = pack2bf(p[(size_t)(2 * v) * ld], p[(size_t)(2 * v + 1) * ld]);
  return u;
}

// ---------------------------------------------------------------------------
// Pack / pad kernels (run once, off the critical path)
// ---------------------------------------------------------------------------
__global__ void pack_w_kernel(const float* __restrict__ W, int ldw, int kLim, int nLim,
                              unsigned short* __restrict__ frag, int nNT) {
  const int kt = blockIdx.x, nt = blockIdx.y;
  const int lane = threadIdx.x & 31;
  const int half = lane >> 4;
  const int col = nt * 16 + (lane & 15);
  const bool cok = col < nLim;
  v8u u;
#pragma unroll
  for (int v = 0; v < 8; ++v) {
    const int k = kt * 32 + half * 16 + 2 * v;
    const float x0 = (cok && k < kLim) ? W[(size_t)k * ldw + col] : 0.f;
    const float x1 = (cok && (k + 1) < kLim) ? W[(size_t)(k + 1) * ldw + col] : 0.f;
    u[v] = pack2bf(x0, x1);
  }
  *(v8u*)(frag + ((size_t)(kt * nNT + nt) * 32 + lane) * 16) = u;
}

__global__ void pad_cache_kernel(const float* __restrict__ cache,
                                 float* __restrict__ cachep) {
  const int idx = blockIdx.x * blockDim.x + threadIdx.x;   // 128*12288 total
  const int q = idx / NTKP;
  const int r = idx - q * NTKP;
  const int t = r / KP;
  const int k = r - t * KP;
  cachep[idx] = (k < NK) ? cache[(size_t)q * NT * NK + t * NK + k] : 0.f;
}

__global__ void pad_bias_kernel(const float* __restrict__ b1, const float* __restrict__ b2,
                                float* __restrict__ b1p, float* __restrict__ b2p) {
  const int i = threadIdx.x;
  if (i < KP) {
    b1p[i] = (i < NK) ? b1[i] : 0.f;
    b2p[i] = (i < NK) ? b2[i] : 0.f;
  }
}

__global__ void zero_kernel(float* __restrict__ p, int n) {
  const int i = blockIdx.x * blockDim.x + threadIdx.x;
  if (i < n) p[i] = 0.f;
}

// ---------------------------------------------------------------------------
// Kernel 1: logitsp = feats @ W1 + b1. One wave = 16 rows x 96 cols (6 tiles).
// W1 fragments double-buffered in LDS via ASYNC global->LDS copies; the wait
// is placed AFTER the WMMA burst so the copy overlaps compute.
// grid 256, block 256 (8 waves).
// ---------------------------------------------------------------------------
__global__ void gemm1_kernel(const float* __restrict__ feats,
                             const unsigned short* __restrict__ w1frag,
                             const float* __restrict__ b1p,
                             float* __restrict__ logitsp) {
  __shared__ uint4 lbuf[2][384];          // 2 x 6 KB (one k-slice: 6 frags x 1KB)
  const int tid = threadIdx.x;
  const int wave = tid >> 5;
  const int lane = tid & 31;
  const int row0 = (blockIdx.x * 8 + wave) * 16;
  const uint4* gw = (const uint4*)w1frag; // 24 * 384 uint4

  // stage slice 0 (async)
  for (int i = tid; i < 384; i += 256) async_g2l_b128(&lbuf[0][i], gw + i);
  wait_async0();
  __syncthreads();

  v8f acc[6] = {};
  v8u a = load_a_frag(feats, ND, row0, 0);
  for (int kt = 0; kt < ND / 32; ++kt) {
    const int cur = kt & 1;
    if (kt < ND / 32 - 1) {
      const uint4* g = gw + (kt + 1) * 384;
      for (int i = tid; i < 384; i += 256) async_g2l_b128(&lbuf[cur ^ 1][i], g + i);
    }
    v8u a_next = a;
    if (kt < ND / 32 - 1) a_next = load_a_frag(feats, ND, row0, (kt + 1) * 32);
#pragma unroll
    for (int nt = 0; nt < 6; ++nt) {
      const uint4* p = &lbuf[cur][nt * 64 + lane * 2];
      acc[nt] = WMMA_BF16(a, frag_from_uint4(p[0], p[1]), acc[nt]);
    }
    a = a_next;
    wait_async0();          // next slice landed (overlapped with WMMAs above)
    __syncthreads();
  }
  const int rb = (lane >> 4) << 3;
#pragma unroll
  for (int nt = 0; nt < 6; ++nt) {
    const int col = nt * 16 + (lane & 15);
    const float bias = b1p[col];
#pragma unroll
    for (int r = 0; r < 8; ++r)
      logitsp[(size_t)(row0 + r + rb) * KP + col] = acc[nt][r] + bias;
  }
}

// ---------------------------------------------------------------------------
// Kernel 2: sim[p,q] = <logitsp[p,:], cachep[q,:]>, split-K accumulate.
// grid 16, block 256. Block -> (32-row group, 64-col group); wave -> k-chunk.
// ---------------------------------------------------------------------------
__global__ void sim_kernel(const float* __restrict__ logitsp,
                           const float* __restrict__ cachep,
                           float* __restrict__ simbuf) {
  const int tid = threadIdx.x;
  const int wave = tid >> 5;
  const int lane = tid & 31;
  const int mrow = (blockIdx.x >> 1) * 32;      // 0..224
  const int ncol = (blockIdx.x & 1) * 64;       // 0 or 64
  const int kbeg = wave * (NTKP / 8);           // 1536-wide chunk
  v8f acc[8] = {};                              // [mi*4 + nj]
  for (int k0 = kbeg; k0 < kbeg + NTKP / 8; k0 += 32) {
    const v8u a0 = load_a_frag(logitsp, NTKP, mrow, k0);
    const v8u a1 = load_a_frag(logitsp, NTKP, mrow + 16, k0);
#pragma unroll
    for (int nj = 0; nj < 4; ++nj) {
      const v8u b = load_bt_frag(cachep, NTKP, k0, ncol + nj * 16);
      acc[nj]     = WMMA_BF16(a0, b, acc[nj]);
      acc[4 + nj] = WMMA_BF16(a1, b, acc[4 + nj]);
    }
  }
  const int rb = (lane >> 4) << 3;
#pragma unroll
  for (int mi = 0; mi < 2; ++mi)
#pragma unroll
    for (int nj = 0; nj < 4; ++nj) {
      const int col = ncol + nj * 16 + (lane & 15);
#pragma unroll
      for (int r = 0; r < 8; ++r)
        atomicAdd(&simbuf[(size_t)(mrow + mi * 16 + r + rb) * NC + col],
                  acc[mi * 4 + nj][r]);
    }
}

// ---------------------------------------------------------------------------
// Kernel 3: softmax over axis=1 (128), in place. grid 256, block 128.
// ---------------------------------------------------------------------------
__global__ void softmax_kernel(float* __restrict__ sim) {
  __shared__ float red[NC];
  const int p = blockIdx.x;
  const int q = threadIdx.x;
  const float v = sim[(size_t)p * NC + q];
  red[q] = v;
  __syncthreads();
  for (int s = 64; s > 0; s >>= 1) {
    if (q < s) red[q] = fmaxf(red[q], red[q + s]);
    __syncthreads();
  }
  const float mx = red[0];
  __syncthreads();
  const float e = expf(v - mx);
  red[q] = e;
  __syncthreads();
  for (int s = 64; s > 0; s >>= 1) {
    if (q < s) red[q] += red[q + s];
    __syncthreads();
  }
  sim[(size_t)p * NC + q] = e / red[0];
}

// ---------------------------------------------------------------------------
// Kernel 4: logitsp += 0.5 * (p @ cachep)  (in place).
// Block packs its 128x128 slice of p into LDS as bf16 A-fragments (32 KB).
// grid 192 (2 row-halves x 96 ntile-groups), block 256.
// ---------------------------------------------------------------------------
__global__ void hgemm_kernel(const float* __restrict__ pmat,
                             const float* __restrict__ cachep,
                             float* __restrict__ logitsp) {
  __shared__ uint4 abuf[2048];            // 4 kt x 8 sub x 32 lanes x 32B = 32 KB
  const int tid = threadIdx.x;
  const int mbase = (blockIdx.x / 96) * 128;
  const int nt = (blockIdx.x % 96) * 8 + (tid >> 5);
  for (int fi = tid; fi < 1024; fi += 256) {
    const int kt = fi >> 8;
    const int sub = (fi >> 5) & 7;
    const int ln = fi & 31;
    const int half = ln >> 4;
    const int m = ln & 15;
    const float* rp = pmat + (size_t)(mbase + sub * 16 + m) * NC + kt * 32 + half * 16;
    const float4 a0 = ((const float4*)rp)[0];
    const float4 a1 = ((const float4*)rp)[1];
    const float4 b0 = ((const float4*)(rp + 8))[0];
    const float4 b1 = ((const float4*)(rp + 8))[1];
    uint4 lo, hi;
    lo.x = pack2bf(a0.x, a0.y); lo.y = pack2bf(a0.z, a0.w);
    lo.z = pack2bf(a1.x, a1.y); lo.w = pack2bf(a1.z, a1.w);
    hi.x = pack2bf(b0.x, b0.y); hi.y = pack2bf(b0.z, b0.w);
    hi.z = pack2bf(b1.x, b1.y); hi.w = pack2bf(b1.z, b1.w);
    abuf[(kt * 8 + sub) * 64 + ln * 2] = lo;
    abuf[(kt * 8 + sub) * 64 + ln * 2 + 1] = hi;
  }
  __syncthreads();
  const int lane = tid & 31;
  v8f acc[8] = {};
#pragma unroll
  for (int kt = 0; kt < 4; ++kt) {
    const v8u b = load_b_frag(cachep, NTKP, kt * 32, nt * 16);
#pragma unroll
    for (int sub = 0; sub < 8; ++sub) {
      const uint4* p = &abuf[(kt * 8 + sub) * 64 + lane * 2];
      acc[sub] = WMMA_BF16(frag_from_uint4(p[0], p[1]), b, acc[sub]);
    }
  }
  const int col = nt * 16 + (lane & 15);
  const int rb = (lane >> 4) << 3;
#pragma unroll
  for (int sub = 0; sub < 8; ++sub)
#pragma unroll
    for (int r = 0; r < 8; ++r) {
      const size_t idx = (size_t)(mbase + sub * 16 + r + rb) * NTKP + col;
      logitsp[idx] = logitsp[idx] + 0.5f * acc[sub][r];
    }
}

// ---------------------------------------------------------------------------
// Kernel 5: newlogp = logitsp(32768x96) @ W2 + b2. W2 frags async-staged in LDS.
// grid 256, block 256.
// ---------------------------------------------------------------------------
__global__ void gemm2_kernel(const float* __restrict__ logitsp,
                             const unsigned short* __restrict__ w2frag,
                             const float* __restrict__ b2p,
                             float* __restrict__ newlogp) {
  __shared__ uint4 wbuf[1152];            // 3 kt x 6 nt x 64 = 18 KB
  const int tid = threadIdx.x;
  const int wave = tid >> 5;
  const int lane = tid & 31;
  const int row0 = (blockIdx.x * 8 + wave) * 16;
  const uint4* gw = (const uint4*)w2frag;
  for (int i = tid; i < 1152; i += 256) async_g2l_b128(&wbuf[i], gw + i);
  // overlap the async copy with the first A fragment load
  v8u a = load_a_frag(logitsp, KP, row0, 0);
  wait_async0();
  __syncthreads();
  v8f acc[6] = {};
#pragma unroll
  for (int kt = 0; kt < 3; ++kt) {
    v8u a_next = a;
    if (kt < 2) a_next = load_a_frag(logitsp, KP, row0, (kt + 1) * 32);
#pragma unroll
    for (int nt = 0; nt < 6; ++nt) {
      const uint4* p = &wbuf[(kt * 6 + nt) * 64 + lane * 2];
      acc[nt] = WMMA_BF16(a, frag_from_uint4(p[0], p[1]), acc[nt]);
    }
    a = a_next;
  }
  const int rb = (lane >> 4) << 3;
#pragma unroll
  for (int nt = 0; nt < 6; ++nt) {
    const int col = nt * 16 + (lane & 15);
    const float bias = b2p[col];
#pragma unroll
    for (int r = 0; r < 8; ++r)
      newlogp[(size_t)(row0 + r + rb) * KP + col] = acc[nt][r] + bias;
  }
}

// ---------------------------------------------------------------------------
// Kernel 6: CRF normalizer. One block (128 thr) per batch; trans in LDS.
// ---------------------------------------------------------------------------
__global__ void crf_fwd_kernel(const float* __restrict__ newlogp,
                               const int* __restrict__ mask,
                               const float* __restrict__ startv,
                               const float* __restrict__ endv,
                               const float* __restrict__ trans,
                               float* __restrict__ den) {
  __shared__ float tr[NK * NK];
  __shared__ float score[80];
  __shared__ float nxt[80];
  const int b = blockIdx.x;
  const int tid = threadIdx.x;
  for (int i = tid; i < NK * NK; i += blockDim.x) tr[i] = trans[i];
  const float* em = newlogp + (size_t)b * NT * KP;
  if (tid < NK) score[tid] = startv[tid] + em[tid];
  __syncthreads();
  for (int t = 1; t < NT; ++t) {
    if (tid < NK) {
      float m = -3.402823466e38f;
      for (int i = 0; i < NK; ++i) m = fmaxf(m, score[i] + tr[i * NK + tid]);
      float s = 0.f;
      for (int i = 0; i < NK; ++i) s += expf(score[i] + tr[i * NK + tid] - m);
      nxt[tid] = m + logf(s) + em[t * KP + tid];
    }
    __syncthreads();
    if (tid < NK && mask[b * NT + t]) score[tid] = nxt[tid];
    __syncthreads();
  }
  if (tid < NK) nxt[tid] = score[tid] + endv[tid];
  __syncthreads();
  if (tid == 0) {
    float m = -3.402823466e38f;
    for (int i = 0; i < NK; ++i) m = fmaxf(m, nxt[i]);
    float s = 0.f;
    for (int i = 0; i < NK; ++i) s += expf(nxt[i] - m);
    den[b] = m + logf(s);
  }
}

// ---------------------------------------------------------------------------
// Kernel 7: CRF gold-path score. One thread per batch.
// ---------------------------------------------------------------------------
__global__ void crf_score_kernel(const float* __restrict__ newlogp,
                                 const int* __restrict__ labels,
                                 const int* __restrict__ mask,
                                 const float* __restrict__ startv,
                                 const float* __restrict__ endv,
                                 const float* __restrict__ trans,
                                 float* __restrict__ num) {
  const int b = blockIdx.x * blockDim.x + threadIdx.x;
  if (b >= NB) return;
  const float* em = newlogp + (size_t)b * NT * KP;
  int prev = labels[b * NT];
  float score = startv[prev] + em[prev];
  int cnt = (mask[b * NT] != 0) ? 1 : 0;
  for (int t = 1; t < NT; ++t) {
    const int l = labels[b * NT + t];
    const float m = (float)mask[b * NT + t];
    score += (trans[prev * NK + l] + em[t * KP + l]) * m;
    prev = l;
    cnt += (mask[b * NT + t] != 0) ? 1 : 0;
  }
  const int last = labels[b * NT + (cnt - 1)];
  num[b] = score + endv[last];
}

// ---------------------------------------------------------------------------
// Kernel 8: Viterbi + backtrack. One block (128 thr) per batch.
// ---------------------------------------------------------------------------
__global__ void viterbi_kernel(const float* __restrict__ newlogp,
                               const int* __restrict__ mask,
                               const float* __restrict__ startv,
                               const float* __restrict__ endv,
                               const float* __restrict__ trans,
                               unsigned char* __restrict__ hist,
                               float* __restrict__ out_tags) {
  __shared__ float tr[NK * NK];
  __shared__ float score[80];
  __shared__ float nxt[80];
  __shared__ int tags[NT];
  const int b = blockIdx.x;
  const int tid = threadIdx.x;
  for (int i = tid; i < NK * NK; i += blockDim.x) tr[i] = trans[i];
  const float* em = newlogp + (size_t)b * NT * KP;
  unsigned char* hb = hist + (size_t)b * (NT - 1) * NK;
  if (tid < NK) score[tid] = startv[tid] + em[tid];
  __syncthreads();
  for (int t = 1; t < NT; ++t) {
    if (tid < NK) {
      float m = -3.402823466e38f;
      int arg = 0;
      for (int i = 0; i < NK; ++i) {
        const float v = score[i] + tr[i * NK + tid];
        if (v > m) { m = v; arg = i; }
      }
      hb[(t - 1) * NK + tid] = (unsigned char)arg;
      nxt[tid] = m + em[t * KP + tid];
    }
    __syncthreads();
    if (tid < NK && mask[b * NT + t]) score[tid] = nxt[tid];
    __syncthreads();
  }
  if (tid == 0) {
    float best = -3.402823466e38f;
    int bl = 0;
    for (int k = 0; k < NK; ++k) {
      const float v = score[k] + endv[k];
      if (v > best) { best = v; bl = k; }
    }
    int cur = bl;
    tags[NT - 1] = bl;
    for (int t = NT - 1; t >= 1; --t) {
      cur = mask[b * NT + t] ? (int)hb[(t - 1) * NK + cur] : bl;
      tags[t - 1] = cur;
    }
  }
  __syncthreads();
  for (int t = tid; t < NT; t += blockDim.x)
    out_tags[(size_t)b * NT + t] = (float)tags[t];
}

// ---------------------------------------------------------------------------
// Kernel 9: loss = -mean(num - den) -> d_out[B*T]
// ---------------------------------------------------------------------------
__global__ void loss_kernel(const float* __restrict__ num,
                            const float* __restrict__ den,
                            float* __restrict__ out) {
  __shared__ float red[NB];
  const int i = threadIdx.x;
  red[i] = num[i] - den[i];
  __syncthreads();
  for (int s = NB / 2; s > 0; s >>= 1) {
    if (i < s) red[i] += red[i + s];
    __syncthreads();
  }
  if (i == 0) out[0] = -red[0] / (float)NB;
}

// ---------------------------------------------------------------------------
extern "C" void kernel_launch(void* const* d_in, const int* in_sizes, int n_in,
                              void* d_out, int out_size, void* d_ws, size_t ws_size,
                              hipStream_t stream) {
  const float* feats  = (const float*)d_in[0];
  const int*   attn   = (const int*)d_in[1];
  const int*   labels = (const int*)d_in[2];
  const float* cache  = (const float*)d_in[3];
  const float* W1     = (const float*)d_in[4];
  const float* b1     = (const float*)d_in[5];
  const float* W2     = (const float*)d_in[6];
  const float* b2     = (const float*)d_in[7];
  const float* startT = (const float*)d_in[8];
  const float* endT   = (const float*)d_in[9];
  const float* trans  = (const float*)d_in[10];
  float* out = (float*)d_out;

  char* ws = (char*)d_ws;
  size_t off = 0;
  auto walloc = [&](size_t bytes) -> void* {
    void* p = ws + off;
    off = (off + bytes + 255) & ~(size_t)255;
    return p;
  };
  float* logitsp = (float*)walloc((size_t)NB * NT * KP * 4);     // 12.6 MB
  float* newlogp = (float*)walloc((size_t)NB * NT * KP * 4);     // 12.6 MB
  float* cachep  = (float*)walloc((size_t)NC * NTKP * 4);        // 6.3 MB
  float* simbuf  = (float*)walloc((size_t)NB * NC * 4);          // 131 KB
  unsigned short* w1frag = (unsigned short*)walloc((size_t)24 * 6 * 512 * 2); // 147 KB
  unsigned short* w2frag = (unsigned short*)walloc((size_t)3 * 6 * 512 * 2);  // 18 KB
  float* b1p  = (float*)walloc(KP * 4);
  float* b2p  = (float*)walloc(KP * 4);
  float* numb = (float*)walloc((size_t)NB * 4);
  float* denb = (float*)walloc((size_t)NB * 4);
  unsigned char* hist = (unsigned char*)walloc((size_t)NB * (NT - 1) * NK);   // 2.4 MB

  // --- prep ---
  pack_w_kernel<<<dim3(24, 6), 32, 0, stream>>>(W1, NK, ND, NK, w1frag, 6);
  pack_w_kernel<<<dim3(3, 6), 32, 0, stream>>>(W2, NK, NK, NK, w2frag, 6);
  pad_cache_kernel<<<(NC * NTKP) / 256, 256, 0, stream>>>(cache, cachep);
  pad_bias_kernel<<<1, 128, 0, stream>>>(b1, b2, b1p, b2p);
  zero_kernel<<<(NB * NC + 255) / 256, 256, 0, stream>>>(simbuf, NB * NC);

  // --- pipeline ---
  gemm1_kernel<<<256, 256, 0, stream>>>(feats, w1frag, b1p, logitsp);
  sim_kernel<<<16, 256, 0, stream>>>(logitsp, cachep, simbuf);
  softmax_kernel<<<NB, NC, 0, stream>>>(simbuf);
  hgemm_kernel<<<192, 256, 0, stream>>>(simbuf, cachep, logitsp);
  gemm2_kernel<<<256, 256, 0, stream>>>(logitsp, w2frag, b2p, newlogp);
  crf_fwd_kernel<<<NB, 128, 0, stream>>>(newlogp, attn, startT, endT, trans, denb);
  crf_score_kernel<<<1, 256, 0, stream>>>(newlogp, labels, attn, startT, endT, trans, numb);
  viterbi_kernel<<<NB, 128, 0, stream>>>(newlogp, attn, startT, endT, trans, hist, out);
  loss_kernel<<<1, NB, 0, stream>>>(numb, denb, out + (size_t)NB * NT);
}
